// ConservativeMPLayer_15917148799343
// MI455X (gfx1250) — matrix operand order repaired and checked by reading, not compiled, not run on hardware
//
#include <hip/hip_runtime.h>
#include <hip/hip_bf16.h>

typedef __attribute__((ext_vector_type(16))) _Float16 v16h;
typedef __attribute__((ext_vector_type(8)))  _Float16 v8h;
typedef __attribute__((ext_vector_type(8)))  float    v8f;

#define NNODES 100000
#define NEDGES 600000

// ---------------- workspace layout (bytes) ----------------
// [0, 4N*4)          : delta_rho (N), delta_e (N), delta_rhou (2N)  -- zeroed each launch
// OFF_BIAS           : 320 floats: bg0cat(64) | bg1sum(64) | bp0cat(192)
// OFF_FRAG           : packed f16 WMMA B-fragments (48128 halves)
// OFF_G              : per-node g features, f16 [N][64]
#define OFF_DRHO   0
#define OFF_DE     (NNODES*4)
#define OFF_DRU    (2*NNODES*4)
#define OFF_BIAS   (4*NNODES*4)
#define OFF_FRAG   (OFF_BIAS + 512*4)
#define OFF_G      (OFF_FRAG + 48128*2)

// fragment sub-offsets (in halves)
#define HF_Fn0 0        // [32(pad from 5),64]  1k x 4n
#define HF_Fn1 2048     // [64,64]              2k x 4n
#define HF_Fg0 6144     // [64,64]=phi1W0|phi2W0 cols   2k x 4n
#define HF_Fg1 10240    // [64,64]=phi1W1 over phi2W1   2k x 4n
#define HF_Fe1 14336    // [32,32]              1k x 2n
#define HF_Fm0 15360    // [96,128]             3k x 8n
#define HF_Fm1 27648    // [128,64]             4k x 4n
#define HF_Fp0 35840    // [64,192]=psi cat     2k x 12n

// Branchless GELU (tanh form) on the native exp2/rcp path:
//   gelu(x) ~= x * sigmoid(1.5957691*(x + 0.044715 x^3))
//   w = x*(c1 + c2*x^2) already in log2 domain; result = x * rcp(1 + exp2(-w))
// Saturates cleanly: w->+inf => x, w->-inf => 0. 2 trans ops, no branches.
__device__ __forceinline__ float gelu_f(float x) {
  float x2 = x * x;
  float w  = x * __builtin_fmaf(0.10294546f, x2, 2.30226426f);
  float e  = __builtin_amdgcn_exp2f(-w);
  return x * __builtin_amdgcn_rcpf(1.0f + e);
}

__device__ __forceinline__ float sigmoid_f(float x) {
  float e = __builtin_amdgcn_exp2f(-1.44269504f * x);
  return __builtin_amdgcn_rcpf(1.0f + e);
}

__device__ __forceinline__ v16h cat8(v8h lo, v8h hi) {
  return __builtin_shufflevector(lo, hi, 0,1,2,3,4,5,6,7,8,9,10,11,12,13,14,15);
}

// A-operand (16x32 f16) fragment from an LDS tile (row-major, strideH halves).
// Lane L<16 holds row M=L, K = kb..kb+7 (VGPR0-3) and kb+16..kb+23 (VGPR4-7),
// lanes 16-31 same rows with kb+8 base (ISA 7.12.2).
__device__ __forceinline__ v16h load_a_frag_lds(const _Float16* tile, int strideH, int kk, int lane) {
  int row = lane & 15;
  int kb  = kk * 32 + ((lane & 16) ? 8 : 0);
  const _Float16* p = tile + row * strideH + kb;
  v8h lo = *(const v8h*)(p);
  v8h hi = *(const v8h*)(p + 16);
  return cat8(lo, hi);
}

// B-operand fragment, pre-packed in global scratch: 16 contiguous halves/lane.
__device__ __forceinline__ v16h load_b_frag(const _Float16* fr, int kStepsTotal, int nt, int kk, int lane) {
  const _Float16* p = fr + (size_t)(((nt * kStepsTotal) + kk) * 32 + lane) * 16;
  v8h lo = *(const v8h*)(p);
  v8h hi = *(const v8h*)(p + 8);
  return cat8(lo, hi);
}

// One wave computes a [16, nTiles*16] = [16,K] @ [K, nTiles*16] GEMM with f32 accum,
// bias add, optional fast GELU, result stored f16 into an LDS tile.
__device__ __forceinline__ void wave_gemm_lds(const _Float16* Atile, int sA,
                                              const _Float16* Bfr, int kSteps, int nTiles,
                                              _Float16* Otile, int sO, int colBase,
                                              const float* bias, bool act, int lane) {
  int cl = lane & 15;
  int rbase = (lane & 16) ? 8 : 0;
  for (int nt = 0; nt < nTiles; ++nt) {
    v8f c = {};
    for (int kk = 0; kk < kSteps; ++kk) {
      v16h a = load_a_frag_lds(Atile, sA, kk, lane);
      v16h b = load_b_frag(Bfr, kSteps, nt, kk, lane);
      c = __builtin_amdgcn_wmma_f32_16x16x32_f16(false, a, false, b, (short)0, c, false, false);
    }
    int col = nt * 16 + cl;
    float bv = bias[col];
#pragma unroll
    for (int i = 0; i < 8; ++i) {
      float x = c[i] + bv;
      if (act) x = gelu_f(x);
      Otile[(rbase + i) * sO + colBase + col] = (_Float16)x;
    }
  }
}

// ---- pack fp32 weight [Krows, Ncols] into f16 WMMA-B fragment layout ----
// B layout per 32-K step: col = nt*16 + (lane&15); K = (lane>=16?8:0) + j2 for j2<8, 16 + (lane>=16?8:0) + (j2-8) for j2>=8.
__global__ void k_pack(const float* __restrict__ W, int Krows, int Ncols,
                       _Float16* __restrict__ dst, int kStepsLocal, int nTilesLocal,
                       int kkBase, int kStepsTotal, int ntBase) {
  int t = blockIdx.x * blockDim.x + threadIdx.x;
  int total = nTilesLocal * kStepsLocal * 512;
  if (t >= total) return;
  int j2   = t & 15;
  int lane = (t >> 4) & 31;
  int fi   = t >> 9;
  int kkL  = fi % kStepsLocal;
  int ntL  = fi / kStepsLocal;
  int col  = ntL * 16 + (lane & 15);
  int K    = kkL * 32 + ((lane >> 4) * 8) + (j2 < 8 ? j2 : 16 + (j2 - 8));
  float v  = (K < Krows) ? W[K * Ncols + col] : 0.0f;
  dst[(size_t)((((ntBase + ntL) * kStepsTotal) + (kkBase + kkL)) * 32 + lane) * 16 + j2] = (_Float16)v;
}

// combined biases: bg0cat(64)=phi1b0|phi2b0, bg1sum(64)=phi1b1+phi2b1, bp0cat(192)=psi b0 cat
__global__ void k_prep_bias(const float* p1b0, const float* p2b0,
                            const float* p1b1, const float* p2b1,
                            const float* prb0, const float* peb0, const float* prub0,
                            float* out) {
  int t = threadIdx.x;
  if (t < 32)       out[t] = p1b0[t];
  else if (t < 64)  out[t] = p2b0[t - 32];
  else if (t < 128) out[t] = p1b1[t - 64] + p2b1[t - 64];
  else if (t < 192) out[t] = prb0[t - 128];
  else if (t < 256) out[t] = peb0[t - 192];
  else if (t < 320) out[t] = prub0[t - 256];
}

__global__ void k_zero(float* p, int n) {
  int i = blockIdx.x * blockDim.x + threadIdx.x;
  if (i < n) p[i] = 0.0f;
}

// ---- node stage: h = phi_node(u);  g = phi1(h)+phi2(h)  (f16 out) ----
__global__ void __launch_bounds__(32) k_node(const float* __restrict__ nu,
                       const _Float16* __restrict__ Fn0, const _Float16* __restrict__ Fn1,
                       const _Float16* __restrict__ Fg0, const _Float16* __restrict__ Fg1,
                       const float* __restrict__ bn0, const float* __restrict__ bn1,
                       const float* __restrict__ bcat,
                       _Float16* __restrict__ g) {
  __shared__ __align__(16) _Float16 X32[16 * 40];
  __shared__ __align__(16) _Float16 T1[16 * 72];
  __shared__ __align__(16) _Float16 T2[16 * 72];
  int lane = threadIdx.x;
  int tile = blockIdx.x;

  if (lane < 16) {
    int node = tile * 16 + lane;
#pragma unroll
    for (int c = 0; c < 32; ++c) {
      float v = (c < 5 && node < NNODES) ? nu[node * 5 + c] : 0.0f;
      X32[lane * 40 + c] = (_Float16)v;
    }
  }
  __syncthreads();
  wave_gemm_lds(X32, 40, Fn0, 1, 4, T1, 72, 0, bn0, true, lane);   // gelu(u@Wn0+b)
  __syncthreads();
  wave_gemm_lds(T1, 72, Fn1, 2, 4, T2, 72, 0, bn1, false, lane);   // h
  __syncthreads();
  wave_gemm_lds(T2, 72, Fg0, 2, 4, T1, 72, 0, bcat, true, lane);   // gelu(h@[W1a|W2a]+bcat)
  __syncthreads();
  {  // g = T1 @ Fg1 + bg1sum -> global f16
    int cl = lane & 15;
    int rbase = (lane & 16) ? 8 : 0;
    for (int nt = 0; nt < 4; ++nt) {
      v8f c = {};
      for (int kk = 0; kk < 2; ++kk) {
        v16h a = load_a_frag_lds(T1, 72, kk, lane);
        v16h b = load_b_frag(Fg1, 2, nt, kk, lane);
        c = __builtin_amdgcn_wmma_f32_16x16x32_f16(false, a, false, b, (short)0, c, false, false);
      }
      int col = nt * 16 + cl;
      float bv = bcat[64 + col];
#pragma unroll
      for (int i = 0; i < 8; ++i) {
        int node = tile * 16 + rbase + i;
        if (node < NNODES) g[(size_t)node * 64 + col] = (_Float16)(c[i] + bv);
      }
    }
  }
}

// ---- edge stage: v=g[src]+g[dst]; eps=phi_edge(r); m=phi_msg([v,eps]); heads; scatter ----
__global__ void __launch_bounds__(32) k_edge(const _Float16* __restrict__ g,
                       const int* __restrict__ eidx, const float* __restrict__ eattr,
                       const float* __restrict__ area,
                       const _Float16* __restrict__ Fe1, const _Float16* __restrict__ Fm0,
                       const _Float16* __restrict__ Fm1, const _Float16* __restrict__ Fp0,
                       const float* __restrict__ we0, const float* __restrict__ be0,
                       const float* __restrict__ be1,
                       const float* __restrict__ bm0, const float* __restrict__ bm1,
                       const float* __restrict__ bp0cat,
                       const float* __restrict__ wrho, const float* __restrict__ brho,
                       const float* __restrict__ wpe,  const float* __restrict__ bpe,
                       const float* __restrict__ wru,  const float* __restrict__ bru,
                       const float* __restrict__ sptr,
                       float* __restrict__ drho, float* __restrict__ de,
                       float* __restrict__ dru) {
  __shared__ __align__(16) _Float16 X96[16 * 104];   // cols 0-63: v, 64-95: eps
  __shared__ __align__(16) _Float16 EH[16 * 40];
  __shared__ __align__(16) _Float16 H128[16 * 136];
  __shared__ __align__(16) _Float16 M64[16 * 72];
  __shared__ __align__(16) _Float16 P192[16 * 200];
  __shared__ int   s_src[16], s_dst[16];
  __shared__ float s_r[16], s_nx[16], s_ny[16];
  __shared__ float s_h[4][16];  // a_rho, a_e, a_rhou0, a_rhou1
  int lane = threadIdx.x;
  int tile = blockIdx.x;
  int row  = lane & 15;
  int hid  = lane >> 4;

  if (lane < 16) {
    int e = tile * 16 + lane;
    s_src[lane] = eidx[e];
    s_dst[lane] = eidx[NEDGES + e];
    float ax = eattr[3 * e + 0], ay = eattr[3 * e + 1], r = eattr[3 * e + 2];
    float inv = __builtin_amdgcn_rcpf(r + 1e-12f);
    s_r[lane] = r; s_nx[lane] = ax * inv; s_ny[lane] = ay * inv;
  }
  __syncthreads();
  {  // gather v = g[src] + g[dst]  (packed f16 adds, 64B per lane)
    int si = s_src[row], di = s_dst[row];
    const v8h* gs = (const v8h*)(g + (size_t)si * 64 + hid * 32);
    const v8h* gd = (const v8h*)(g + (size_t)di * 64 + hid * 32);
    v8h* o = (v8h*)(X96 + row * 104 + hid * 32);
#pragma unroll
    for (int q = 0; q < 4; ++q) o[q] = gs[q] + gd[q];
  }
  {  // phi_edge layer 1 is rank-1: hidden = gelu(r*w + b)   (VALU)
    float r = s_r[row];
    int jb = hid * 16;
#pragma unroll
    for (int jj = 0; jj < 16; ++jj) {
      int j = jb + jj;
      EH[row * 40 + j] = (_Float16)gelu_f(r * we0[j] + be0[j]);
    }
  }
  __syncthreads();
  wave_gemm_lds(EH, 40, Fe1, 1, 2, X96, 104, 64, be1, false, lane);   // eps -> X96[:,64:96]
  __syncthreads();
  wave_gemm_lds(X96, 104, Fm0, 3, 8, H128, 136, 0, bm0, true, lane);  // phi_msg L1 (K=96)
  __syncthreads();
  wave_gemm_lds(H128, 136, Fm1, 4, 4, M64, 72, 0, bm1, false, lane);  // phi_msg L2 -> m
  __syncthreads();
  wave_gemm_lds(M64, 72, Fp0, 2, 12, P192, 200, 0, bp0cat, true, lane); // psi hidden (cat 192)
  __syncthreads();
  if (hid == 0) {  // a_rho, a_e head dots
    float s0 = 0.f, s1 = 0.f;
    for (int j = 0; j < 64; ++j) {
      s0 += (float)P192[row * 200 + j]      * wrho[j];
      s1 += (float)P192[row * 200 + 64 + j] * wpe[j];
    }
    s_h[0][row] = s0 + brho[0];
    s_h[1][row] = s1 + bpe[0];
  } else {         // a_rhou head dots
    float s0 = 0.f, s1 = 0.f;
    for (int j = 0; j < 64; ++j) {
      float hv = (float)P192[row * 200 + 128 + j];
      s0 += hv * wru[2 * j + 0];
      s1 += hv * wru[2 * j + 1];
    }
    s_h[2][row] = s0 + bru[0];
    s_h[3][row] = s1 + bru[1];
  }
  __syncthreads();
  {  // flux + conservative scatter: lanes 0-15 = src side, 16-31 = dst side
    int si = s_src[row], di = s_dst[row];
    float mask = (si < di) ? 1.0f : 0.0f;
    float r = s_r[row], nx = s_nx[row], ny = s_ny[row];
    float nn = nx * nx + ny * ny;
    float arho = s_h[0][row], ae = s_h[1][row], am0 = s_h[2][row], am1 = s_h[3][row];
    float rm = r * mask;
    float r_raw = arho * nn * rm;
    float e_raw = ae * nn * rm;
    float m0 = (am0 * nx - am1 * ny) * rm;   // t = (-ny, nx)
    float m1 = (am0 * ny + am1 * nx) * rm;
    float dt = 0.015f * sigmoid_f(sptr[0]);
    int node = hid ? di : si;
    float f = (hid ? 1.0f : -1.0f) * dt * __builtin_amdgcn_rcpf(area[node]);
    atomicAdd(&drho[node], r_raw * f);
    atomicAdd(&de[node],   e_raw * f);
    atomicAdd(&dru[2 * node + 0], m0 * f);
    atomicAdd(&dru[2 * node + 1], m1 * f);
  }
}

// ---- final pointwise update + EOS ----
__global__ void k_final(const float* __restrict__ nu, const float* __restrict__ drho,
                        const float* __restrict__ de, const float* __restrict__ dru,
                        const float* __restrict__ sptr, float* __restrict__ out) {
  int i = blockIdx.x * blockDim.x + threadIdx.x;
  if (i == 0) {
    out[NNODES * 5] = 0.015f * sigmoid_f(sptr[0]);
  }
  if (i >= NNODES) return;
  float rho_n = nu[i * 5 + 0] + drho[i];
  float e_n   = nu[i * 5 + 1] + de[i];
  float ru0   = nu[i * 5 + 3] + dru[2 * i + 0];
  float ru1   = nu[i * 5 + 4] + dru[2 * i + 1];
  float rho_d = rho_n * 0.1f + 1.0f;
  float e_d   = e_n * 0.2f + 2.0f;
  float rd0 = ru0 * 0.2f, rd1 = ru1 * 0.2f;
  float p  = 0.4f * (e_d - 0.5f * (rd0 * rd0 + rd1 * rd1) * __builtin_amdgcn_rcpf(rho_d));
  float pn = (p - 1.0f) * (1.0f / 0.3f);
  out[i * 5 + 0] = rho_n;
  out[i * 5 + 1] = e_n;
  out[i * 5 + 2] = pn;
  out[i * 5 + 3] = ru0;
  out[i * 5 + 4] = ru1;
}

extern "C" void kernel_launch(void* const* d_in, const int* in_sizes, int n_in,
                              void* d_out, int out_size, void* d_ws, size_t ws_size,
                              hipStream_t stream) {
  (void)in_sizes; (void)n_in; (void)out_size; (void)ws_size;
  // inputs, setup_inputs() insertion order; params leaves in dict/tuple order
  const float* nu    = (const float*)d_in[0];
  const int*   eidx  = (const int*)  d_in[1];
  const float* eattr = (const float*)d_in[2];
  const float* area  = (const float*)d_in[3];
  const float* WnW0 = (const float*)d_in[4],  *WnB0 = (const float*)d_in[5];
  const float* WnW1 = (const float*)d_in[6],  *WnB1 = (const float*)d_in[7];
  const float* P1W0 = (const float*)d_in[8],  *P1B0 = (const float*)d_in[9];
  const float* P1W1 = (const float*)d_in[10], *P1B1 = (const float*)d_in[11];
  const float* P2W0 = (const float*)d_in[12], *P2B0 = (const float*)d_in[13];
  const float* P2W1 = (const float*)d_in[14], *P2B1 = (const float*)d_in[15];
  const float* PeW0 = (const float*)d_in[16], *PeB0 = (const float*)d_in[17];
  const float* PeW1 = (const float*)d_in[18], *PeB1 = (const float*)d_in[19];
  const float* PmW0 = (const float*)d_in[20], *PmB0 = (const float*)d_in[21];
  const float* PmW1 = (const float*)d_in[22], *PmB1 = (const float*)d_in[23];
  const float* PrW0 = (const float*)d_in[24], *PrB0 = (const float*)d_in[25];
  const float* PrW1 = (const float*)d_in[26], *PrB1 = (const float*)d_in[27];
  const float* PEW0 = (const float*)d_in[28], *PEB0 = (const float*)d_in[29];
  const float* PEW1 = (const float*)d_in[30], *PEB1 = (const float*)d_in[31];
  const float* PuW0 = (const float*)d_in[32], *PuB0 = (const float*)d_in[33];
  const float* PuW1 = (const float*)d_in[34], *PuB1 = (const float*)d_in[35];
  const float* sptr = (const float*)d_in[36];

  char* ws = (char*)d_ws;
  float*     drho  = (float*)(ws + OFF_DRHO);
  float*     de    = (float*)(ws + OFF_DE);
  float*     dru   = (float*)(ws + OFF_DRU);
  float*     bcat  = (float*)(ws + OFF_BIAS);
  _Float16*  frags = (_Float16*)(ws + OFF_FRAG);
  _Float16*  g     = (_Float16*)(ws + OFF_G);
  float*     outp  = (float*)d_out;

  // deterministic per-call init of accumulators
  k_zero<<<(4 * NNODES + 255) / 256, 256, 0, stream>>>((float*)(ws + OFF_DRHO), 4 * NNODES);
  k_prep_bias<<<1, 320, 0, stream>>>(P1B0, P2B0, P1B1, P2B1, PrB0, PEB0, PuB0, bcat);

  auto pack = [&](const float* W, int Kr, int Nc, _Float16* dst,
                  int kL, int nL, int kkB, int kT, int ntB) {
    int total = nL * kL * 512;
    k_pack<<<(total + 255) / 256, 256, 0, stream>>>(W, Kr, Nc, dst, kL, nL, kkB, kT, ntB);
  };
  pack(WnW0,   5,  64, frags + HF_Fn0, 1, 4, 0, 1, 0);   // phi_node L1 (K padded 5->32)
  pack(WnW1,  64,  64, frags + HF_Fn1, 2, 4, 0, 2, 0);   // phi_node L2
  pack(P1W0,  64,  32, frags + HF_Fg0, 2, 2, 0, 2, 0);   // g L1 = [phi1W0 | phi2W0]
  pack(P2W0,  64,  32, frags + HF_Fg0, 2, 2, 0, 2, 2);
  pack(P1W1,  32,  64, frags + HF_Fg1, 1, 4, 0, 2, 0);   // g L2 = [phi1W1 ; phi2W1]
  pack(P2W1,  32,  64, frags + HF_Fg1, 1, 4, 1, 2, 0);
  pack(PeW1,  32,  32, frags + HF_Fe1, 1, 2, 0, 1, 0);   // phi_edge L2
  pack(PmW0,  96, 128, frags + HF_Fm0, 3, 8, 0, 3, 0);   // phi_msg L1
  pack(PmW1, 128,  64, frags + HF_Fm1, 4, 4, 0, 4, 0);   // phi_msg L2
  pack(PrW0,  64,  64, frags + HF_Fp0, 2, 4, 0, 2, 0);   // psi cat first layer
  pack(PEW0,  64,  64, frags + HF_Fp0, 2, 4, 0, 2, 4);
  pack(PuW0,  64,  64, frags + HF_Fp0, 2, 4, 0, 2, 8);

  k_node<<<(NNODES + 15) / 16, 32, 0, stream>>>(nu,
      frags + HF_Fn0, frags + HF_Fn1, frags + HF_Fg0, frags + HF_Fg1,
      WnB0, WnB1, bcat, g);

  k_edge<<<NEDGES / 16, 32, 0, stream>>>(g, eidx, eattr, area,
      frags + HF_Fe1, frags + HF_Fm0, frags + HF_Fm1, frags + HF_Fp0,
      PeW0, PeB0, PeB1, PmB0, PmB1, bcat + 128,
      PrW1, PrB1, PEW1, PEB1, PuW1, PuB1, sptr,
      drho, de, dru);

  k_final<<<(NNODES + 255) / 256, 256, 0, stream>>>(nu, drho, de, dru, sptr, outp);
}
